// Recurrence_45586782879988
// MI455X (gfx1250) — compile-verified, matrix operand order checked by reading, not run on hardware
//
#include <hip/hip_runtime.h>
#include <hip/hip_bf16.h>
#include <stdint.h>

// ---------- types for WMMA ----------
typedef __bf16 bf16_t;
typedef __attribute__((ext_vector_type(16))) __bf16 bf16x16;
typedef __attribute__((ext_vector_type(8)))  __bf16 bf16x8;
typedef __attribute__((ext_vector_type(8)))  float  f32x8;

// ---------- problem constants ----------
#define T_DIM 128
#define N_DIM 256
#define H_DIM 512
#define C_DIM 64
#define L_DIM 64
#define NA_DIM 64
#define IN_W  130   // C + L + 2
#define OUT_W 643   // 3 + H + 2*NA
#define G3H   1536  // 3*H
#define X0_W  576   // C + H
#define H2    1024  // 2*H

// 16-bit A/B fragment (ISA 7.12.2):
// lane<16  : row r=lane,    K = {0..7, 16..23}
// lane>=16 : row r=lane-16, K = {8..15, 24..31}
__device__ __forceinline__ bf16x16 load_frag16(const bf16_t* __restrict__ base, long ld) {
  int lane = threadIdx.x & 31;
  int half = lane >> 4;
  int r    = lane & 15;
  const bf16_t* p = base + (long)r * ld + half * 8;
  bf16x8 lo = *(const bf16x8*)(p);        // 16B aligned: ld, k0 multiples of 8
  bf16x8 hi = *(const bf16x8*)(p + 16);
  return __builtin_shufflevector(lo, hi, 0,1,2,3,4,5,6,7,8,9,10,11,12,13,14,15);
}

#define WMMA_BF16(A_, B_, C_) \
  __builtin_amdgcn_wmma_f32_16x16x32_bf16(false, (A_), false, (B_), (short)0, (C_), false, false)

// =====================================================================
// WMMA bf16 GEMM, 32x32 macro-tile per wave (2x2 register blocking):
//   C[m, colOff+n] (+= bias[n]) = A(M x K) @ B(N x K)^T
// A: bf16 row-major (lda, aBatch); B: bf16 row-major out x K (ldb==K, bBatch)
// C: f32 or bf16 (ldc, cBatch). 4 WMMAs / 4 fragment loads per K-step.
// =====================================================================
template<bool RELU, bool BF16OUT>
__global__ __launch_bounds__(128)
void wmma_gemm(const bf16_t* __restrict__ A, long lda, long aBatch,
               const bf16_t* __restrict__ B, long bBatch,
               const float*  __restrict__ bias,
               void* __restrict__ Cv, long ldc, long cBatch, int colOff,
               int K, int mtilesN)
{
  const int wave = threadIdx.x >> 5;
  const int lane = threadIdx.x & 31;
  const int mtn = blockIdx.x * 4 + wave;
  if (mtn >= mtilesN) return;
  const int mtm = blockIdx.y;
  const long b = blockIdx.z;

  const bf16_t* Ab = A + b * aBatch + (long)(mtm * 32) * lda;
  const bf16_t* Bb = B + b * bBatch + (long)(mtn * 32) * (long)K;

  f32x8 acc00 = {0.f,0.f,0.f,0.f,0.f,0.f,0.f,0.f};
  f32x8 acc01 = acc00, acc10 = acc00, acc11 = acc00;
  for (int k0 = 0; k0 < K; k0 += 32) {
    bf16x16 a0 = load_frag16(Ab + k0, lda);
    bf16x16 a1 = load_frag16(Ab + (long)16 * lda + k0, lda);
    bf16x16 b0 = load_frag16(Bb + k0, (long)K);
    bf16x16 b1 = load_frag16(Bb + (long)16 * K + k0, (long)K);
    acc00 = WMMA_BF16(a0, b0, acc00);
    acc01 = WMMA_BF16(a0, b1, acc01);
    acc10 = WMMA_BF16(a1, b0, acc10);
    acc11 = WMMA_BF16(a1, b1, acc11);
  }

  // C/D tile layout: VGPR i -> (M = rowBase + i, N = col)
  const int half = lane >> 4;
  const int r    = lane & 15;
  const int col0 = mtn * 32 + r;
  const int col1 = col0 + 16;
  const float bv0 = bias ? bias[col0] : 0.f;
  const float bv1 = bias ? bias[col1] : 0.f;
  const long row0 = (long)(mtm * 32 + half * 8);
  const long row1 = row0 + 16;
  #pragma unroll
  for (int i = 0; i < 8; ++i) {
    float v00 = acc00[i] + bv0, v01 = acc01[i] + bv1;
    float v10 = acc10[i] + bv0, v11 = acc11[i] + bv1;
    if (RELU) {
      v00 = v00 > 0.f ? v00 : 0.f; v01 = v01 > 0.f ? v01 : 0.f;
      v10 = v10 > 0.f ? v10 : 0.f; v11 = v11 > 0.f ? v11 : 0.f;
    }
    if (BF16OUT) {
      bf16_t* C = (bf16_t*)Cv + b * cBatch + colOff;
      C[(row0 + i) * ldc + col0] = (bf16_t)v00;
      C[(row0 + i) * ldc + col1] = (bf16_t)v01;
      C[(row1 + i) * ldc + col0] = (bf16_t)v10;
      C[(row1 + i) * ldc + col1] = (bf16_t)v11;
    } else {
      float* C = (float*)Cv + b * cBatch + colOff;
      C[(row0 + i) * ldc + col0] = v00;
      C[(row0 + i) * ldc + col1] = v01;
      C[(row1 + i) * ldc + col0] = v10;
      C[(row1 + i) * ldc + col1] = v11;
    }
  }
}

// =====================================================================
// Fused GRU step (critical path): one kernel per time step.
// Each of 16 WGs (1024 thr = 32 waves) owns rows [n0, n0+16) of h.
// Phase 1: gh strip (16 x 1536) = h_bf[n0:16] @ Wh^T + bh into LDS
//          (3 column-tiles per wave; A fragment loaded once per K-step,
//           reused across 3 WMMAs).
// Phase 2: barrier, then gate combine in place. Each WG touches only its
//          own 16 rows of h/h_bf -> no cross-WG hazard.
// =====================================================================
__global__ __launch_bounds__(1024)
void gru_step_fused(const bf16_t* __restrict__ Wh,   // 1536 x 512 bf16
                    const float*  __restrict__ bh,   // 1536
                    const float*  __restrict__ gi,   // N x 1536 (this step, incl bi)
                    float*  __restrict__ h,          // N x 512 f32 state
                    bf16_t* __restrict__ h_bf,       // N x 512 bf16 state
                    bf16_t* __restrict__ bfDest, long bfStride,
                    float*  __restrict__ fDest,  long fStride)
{
  extern __shared__ float ghs[];                     // 16 x 1536 f32 = 96 KB
  const int tid  = threadIdx.x;
  const int wave = tid >> 5;
  const int lane = tid & 31;
  const int n0   = blockIdx.x * 16;

  const bf16_t* Ab = h_bf + (long)n0 * H_DIM;
  const int tn0 = wave * 3;                          // 32 waves x 3 tiles = 96
  const bf16_t* Bb = Wh + (long)(tn0 * 16) * H_DIM;

  f32x8 acc0 = {0.f,0.f,0.f,0.f,0.f,0.f,0.f,0.f};
  f32x8 acc1 = acc0, acc2 = acc0;
  for (int k0 = 0; k0 < H_DIM; k0 += 32) {
    bf16x16 a  = load_frag16(Ab + k0, H_DIM);
    bf16x16 b0 = load_frag16(Bb + k0, H_DIM);
    bf16x16 b1 = load_frag16(Bb + (long)16 * H_DIM + k0, H_DIM);
    bf16x16 b2 = load_frag16(Bb + (long)32 * H_DIM + k0, H_DIM);
    acc0 = WMMA_BF16(a, b0, acc0);
    acc1 = WMMA_BF16(a, b1, acc1);
    acc2 = WMMA_BF16(a, b2, acc2);
  }
  const int half = lane >> 4;
  const int r    = lane & 15;
  {
    const int c0 = tn0 * 16 + r, c1 = c0 + 16, c2 = c0 + 32;
    const float bv0 = bh[c0], bv1 = bh[c1], bv2 = bh[c2];
    #pragma unroll
    for (int i = 0; i < 8; ++i) {
      int row = half * 8 + i;
      ghs[row * G3H + c0] = acc0[i] + bv0;
      ghs[row * G3H + c1] = acc1[i] + bv1;
      ghs[row * G3H + c2] = acc2[i] + bv2;
    }
  }
  __syncthreads();

  // gate combine: 16*512 elements, 1024 threads -> 8 each
  for (int e = tid; e < 16 * H_DIM; e += 1024) {
    int rr = e >> 9;
    int j  = e & (H_DIM - 1);
    int n  = n0 + rr;
    long g = (long)n * G3H + j;
    float rg = 1.f / (1.f + expf(-(gi[g]              + ghs[rr * G3H + j])));
    float zg = 1.f / (1.f + expf(-(gi[g + H_DIM]      + ghs[rr * G3H + j + H_DIM])));
    float nn = tanhf(gi[g + 2 * H_DIM] + rg * ghs[rr * G3H + j + 2 * H_DIM]);
    long hi = (long)n * H_DIM + j;
    float hv = (1.f - zg) * nn + zg * h[hi];
    h[hi]    = hv;
    h_bf[hi] = (bf16_t)hv;
    if (bfDest) bfDest[(long)n * bfStride + j] = (bf16_t)hv;
    if (fDest)  fDest[(long)n * fStride + j]  = hv;
  }
}

// =====================================================================
// elementwise / glue kernels
// =====================================================================
__global__ void cvt_to_bf16(const float* __restrict__ s, bf16_t* __restrict__ d, long n) {
  long i = (long)blockIdx.x * blockDim.x + threadIdx.x;
  if (i < n) d[i] = (bf16_t)s[i];
}

__global__ void zero_u32(uint32_t* __restrict__ p, long n) {
  long i = (long)blockIdx.x * blockDim.x + threadIdx.x;
  if (i < n) p[i] = 0u;
}

// p_arr[n] = all(hx[n]==0) ? 0 : (int)hx[n,1];   h = hx[n,3:3+H] (f32 + bf16)
__global__ void hx_init(const float* __restrict__ hx, int* __restrict__ p_arr,
                        float* __restrict__ h, bf16_t* __restrict__ h_bf)
{
  int wave = threadIdx.x >> 5, lane = threadIdx.x & 31;
  int n = blockIdx.x * (blockDim.x >> 5) + wave;
  if (n >= N_DIM) return;
  const float* row = hx + (long)n * OUT_W;
  int nz = 0;
  for (int i = lane; i < OUT_W; i += 32) nz |= (row[i] != 0.f);
  for (int m = 16; m >= 1; m >>= 1) nz |= __shfl_xor(nz, m, 32);
  if (lane == 0) p_arr[n] = nz ? (int)row[1] : 0;
  for (int j = lane; j < H_DIM; j += 32) {
    float v = row[3 + j];
    h[(long)n * H_DIM + j]    = v;
    h_bf[(long)n * H_DIM + j] = (bf16_t)v;
  }
}

// xs[l,n,:] = bf16(emb[(int)inputs[0,n,C+l]])
__global__ void build_xs(const float* __restrict__ inputs, const float* __restrict__ emb,
                         bf16_t* __restrict__ xs)
{
  long idx = (long)blockIdx.x * blockDim.x + threadIdx.x;  // L*N*H
  int j = (int)(idx & (H_DIM - 1));
  long rem = idx >> 9;
  int n = (int)(rem & (N_DIM - 1));
  int l = (int)(rem >> 8);
  int line = (int)inputs[(long)n * IN_W + C_DIM + l];
  xs[idx] = (bf16_t)emb[(long)line * H_DIM + j];
}

// X0[t*N+n, :] = [ cond[t,n, 0..63], emb[lines0[n, p[n]]][0..511] ]  (bf16)
__global__ void build_x0(const float* __restrict__ inputs, const float* __restrict__ emb,
                         const int* __restrict__ p_arr, bf16_t* __restrict__ X0)
{
  long idx = (long)blockIdx.x * blockDim.x + threadIdx.x;  // T*N*576
  int c = (int)(idx % X0_W);
  long row = idx / X0_W;               // t*N + n
  int n = (int)(row & (N_DIM - 1));
  float v;
  if (c < C_DIM) {
    v = inputs[row * IN_W + c];
  } else {
    int line = (int)inputs[(long)n * IN_W + C_DIM + p_arr[n]];
    v = emb[(long)line * H_DIM + (c - C_DIM)];
  }
  X0[idx] = (bf16_t)v;
}

// in-place softmax over 64 contiguous cols of d_out rows (width OUT_W)
__global__ void softmax64(float* __restrict__ out, int colOff)
{
  int wave = threadIdx.x >> 5, lane = threadIdx.x & 31;
  long row = (long)blockIdx.x * (blockDim.x >> 5) + wave;
  if (row >= (long)T_DIM * N_DIM) return;
  float* p = out + row * OUT_W + colOff;
  float x0 = p[lane], x1 = p[lane + 32];
  float m = fmaxf(x0, x1);
  for (int s = 16; s >= 1; s >>= 1) m = fmaxf(m, __shfl_xor(m, s, 32));
  float e0 = expf(x0 - m), e1 = expf(x1 - m);
  float sm = e0 + e1;
  for (int s = 16; s >= 1; s >>= 1) sm += __shfl_xor(sm, s, 32);
  p[lane] = e0 / sm; p[lane + 32] = e1 / sm;
}

// v = h . Wc + bc  (h read from d_out's own f32 h-slot, contiguous per row)
__global__ void value_head(const float* __restrict__ Wc, const float* __restrict__ bc,
                           float* __restrict__ out)
{
  int wave = threadIdx.x >> 5, lane = threadIdx.x & 31;
  long row = (long)blockIdx.x * (blockDim.x >> 5) + wave;
  if (row >= (long)T_DIM * N_DIM) return;
  const float* hrow = out + row * OUT_W + 3;
  float s = 0.f;
  for (int j = lane; j < H_DIM; j += 32) s += hrow[j] * Wc[j];
  for (int m = 16; m >= 1; m >>= 1) s += __shfl_xor(s, m, 32);
  if (lane == 0) out[row * OUT_W + 2] = s + bc[0];
}

__global__ void ap_copy(const float* __restrict__ inputs, float* __restrict__ out)
{
  long row = (long)blockIdx.x * blockDim.x + threadIdx.x;  // T*N
  if (row >= (long)T_DIM * N_DIM) return;
  out[row * OUT_W + 0] = (float)(int)inputs[row * IN_W + C_DIM + L_DIM];
  out[row * OUT_W + 1] = (float)(int)inputs[row * IN_W + C_DIM + L_DIM + 1];
}

// =====================================================================
// host side
// =====================================================================
extern "C" void kernel_launch(void* const* d_in, const int* in_sizes, int n_in,
                              void* d_out, int out_size, void* d_ws, size_t ws_size,
                              hipStream_t stream)
{
  const float* inputs = (const float*)d_in[0];
  const float* hx     = (const float*)d_in[1];
  const float* emb    = (const float*)d_in[2];
  const float* Wi_f = (const float*)d_in[3];  const float* Wh_f = (const float*)d_in[4];
  const float* bi_f = (const float*)d_in[5];  const float* bh_f = (const float*)d_in[6];
  const float* Wi_b = (const float*)d_in[7];  const float* Wh_b = (const float*)d_in[8];
  const float* bi_b = (const float*)d_in[9];  const float* bh_b = (const float*)d_in[10];
  const float* W0 = (const float*)d_in[11];   const float* b0 = (const float*)d_in[12];
  const float* W1 = (const float*)d_in[13];   const float* b1 = (const float*)d_in[14];
  const float* W2 = (const float*)d_in[15];   const float* b2 = (const float*)d_in[16];
  const float* Wi_c = (const float*)d_in[17]; const float* Wh_c = (const float*)d_in[18];
  const float* bi_c = (const float*)d_in[19]; const float* bh_c = (const float*)d_in[20];
  const float* Wa = (const float*)d_in[21];   const float* ba = (const float*)d_in[22];
  const float* Wc = (const float*)d_in[23];   const float* bc = (const float*)d_in[24];
  const float* Wl = (const float*)d_in[25];   const float* bl = (const float*)d_in[26];
  float* out = (float*)d_out;

  // ---- workspace bump allocator (~470 MB total) ----
  char* wsp = (char*)d_ws; size_t off = 0;
  auto alloc = [&](size_t bytes) -> void* {
    off = (off + 255) & ~(size_t)255;
    void* p = wsp + off; off += bytes; return p;
  };
  bf16_t* wWi_f = (bf16_t*)alloc(G3H*H_DIM*2);  bf16_t* wWh_f = (bf16_t*)alloc(G3H*H_DIM*2);
  bf16_t* wWi_b = (bf16_t*)alloc(G3H*H_DIM*2);  bf16_t* wWh_b = (bf16_t*)alloc(G3H*H_DIM*2);
  bf16_t* wWi_c = (bf16_t*)alloc(G3H*H_DIM*2);  bf16_t* wWh_c = (bf16_t*)alloc(G3H*H_DIM*2);
  bf16_t* wW0 = (bf16_t*)alloc((size_t)H_DIM*X0_W*2);
  bf16_t* wW1 = (bf16_t*)alloc((size_t)H_DIM*H_DIM*2);
  bf16_t* wW2 = (bf16_t*)alloc((size_t)H_DIM*H_DIM*2);
  bf16_t* wWa = (bf16_t*)alloc((size_t)NA_DIM*H_DIM*2);
  bf16_t* wWl = (bf16_t*)alloc((size_t)H2*H_DIM*2);
  bf16_t* xs  = (bf16_t*)alloc((size_t)L_DIM*N_DIM*H_DIM*2);
  bf16_t* X0b = (bf16_t*)alloc((size_t)T_DIM*N_DIM*X0_W*2);
  bf16_t* Xp  = (bf16_t*)alloc((size_t)T_DIM*N_DIM*H_DIM*2);
  bf16_t* Xq  = (bf16_t*)alloc((size_t)T_DIM*N_DIM*H_DIM*2);
  float*  Gi  = (float*) alloc((size_t)T_DIM*N_DIM*G3H*4);   // reused by both GRU phases
  float*  h_f = (float*) alloc((size_t)N_DIM*H_DIM*4);
  bf16_t* h_bf= (bf16_t*)alloc((size_t)N_DIM*H_DIM*2);
  bf16_t* Kbf = (bf16_t*)alloc((size_t)N_DIM*L_DIM*H2*2);    // K (n, l, 2H)
  bf16_t* Hsb = (bf16_t*)alloc((size_t)T_DIM*N_DIM*H_DIM*2);
  bf16_t* Qs  = (bf16_t*)alloc((size_t)T_DIM*N_DIM*H2*2);
  int*    p_arr = (int*) alloc(N_DIM*4);

  auto cv = [&](const float* s, bf16_t* d, long n) {
    cvt_to_bf16<<<(int)((n + 255) / 256), 256, 0, stream>>>(s, d, n);
  };
  // macro-tile counts: mtM = M/32, mtN = Nout/32
  auto gemm = [&](bool relu, bool bf16out,
                  const bf16_t* A, long lda, long aBatch,
                  const bf16_t* B, long bBatch, const float* bias,
                  void* C, long ldc, long cBatch, int colOff,
                  int K, int mtM, int mtN, int batches) {
    dim3 grid((mtN + 3) / 4, mtM, batches), blk(128);
    if (bf16out) {
      if (relu) wmma_gemm<true ,true ><<<grid, blk, 0, stream>>>(A,lda,aBatch,B,bBatch,bias,C,ldc,cBatch,colOff,K,mtN);
      else      wmma_gemm<false,true ><<<grid, blk, 0, stream>>>(A,lda,aBatch,B,bBatch,bias,C,ldc,cBatch,colOff,K,mtN);
    } else {
      if (relu) wmma_gemm<true ,false><<<grid, blk, 0, stream>>>(A,lda,aBatch,B,bBatch,bias,C,ldc,cBatch,colOff,K,mtN);
      else      wmma_gemm<false,false><<<grid, blk, 0, stream>>>(A,lda,aBatch,B,bBatch,bias,C,ldc,cBatch,colOff,K,mtN);
    }
  };
  const size_t GH_LDS = (size_t)16 * G3H * sizeof(float);    // 96 KB dynamic LDS
  auto gru_step = [&](const bf16_t* Wh, const float* bh, const float* gi_t,
                      bf16_t* bfDest, long bfStride, float* fDest, long fStride) {
    gru_step_fused<<<dim3(N_DIM / 16), dim3(1024), GH_LDS, stream>>>(
        Wh, bh, gi_t, h_f, h_bf, bfDest, bfStride, fDest, fStride);
  };
  auto zero_h = [&]() {
    zero_u32<<<512, 256, 0, stream>>>((uint32_t*)h_f, (long)N_DIM * H_DIM);
    zero_u32<<<256, 256, 0, stream>>>((uint32_t*)h_bf, (long)N_DIM * H_DIM / 2);
  };

  // ---- weight conversion to bf16 (L2-resident) ----
  cv(Wi_f, wWi_f, G3H*H_DIM); cv(Wh_f, wWh_f, G3H*H_DIM);
  cv(Wi_b, wWi_b, G3H*H_DIM); cv(Wh_b, wWh_b, G3H*H_DIM);
  cv(Wi_c, wWi_c, G3H*H_DIM); cv(Wh_c, wWh_c, G3H*H_DIM);
  cv(W0, wW0, (long)H_DIM*X0_W); cv(W1, wW1, (long)H_DIM*H_DIM); cv(W2, wW2, (long)H_DIM*H_DIM);
  cv(Wa, wWa, (long)NA_DIM*H_DIM); cv(Wl, wWl, (long)H2*H_DIM);

  // ---- p_arr from hx (needed for X0's r rows) ----
  hx_init<<<N_DIM/8, 256, 0, stream>>>(hx, p_arr, h_f, h_bf);

  // ---- embedding -> xs (L, N, H) bf16 ----
  build_xs<<<(L_DIM*N_DIM*H_DIM)/256, 256, 0, stream>>>(inputs, emb, xs);

  // ================= forward GRU (only gh is sequential) ================
  gemm(false,false, xs, H_DIM, 0, wWi_f, 0, bi_f, Gi, G3H, 0, 0, H_DIM, (L_DIM*N_DIM)/32, 48, 1);
  zero_h();
  for (int s = 0; s < L_DIM; ++s)
    gru_step(wWh_f, bh_f, Gi + (long)s*N_DIM*G3H, Kbf + (long)s*H2, (long)L_DIM*H2, nullptr, 0);

  // ================= backward GRU =======================================
  gemm(false,false, xs, H_DIM, 0, wWi_b, 0, bi_b, Gi, G3H, 0, 0, H_DIM, (L_DIM*N_DIM)/32, 48, 1);
  zero_h();
  for (int s = 0; s < L_DIM; ++s) {
    int l = L_DIM - 1 - s;  // reversed scan, un-reversed storage
    gru_step(wWh_b, bh_b, Gi + (long)l*N_DIM*G3H, Kbf + (long)l*H2 + H_DIM, (long)L_DIM*H2, nullptr, 0);
  }

  // ========== batched MLP front-end + input gates for all T steps =======
  build_x0<<<(int)(((long)T_DIM*N_DIM*X0_W)/256), 256, 0, stream>>>(inputs, emb, p_arr, X0b);
  gemm(true ,true , X0b, X0_W, 0, wW0, 0, b0, Xp, H_DIM, 0, 0, X0_W, (T_DIM*N_DIM)/32, H_DIM/32, 1);
  gemm(true ,true , Xp,  H_DIM,0, wW1, 0, b1, Xq, H_DIM, 0, 0, H_DIM, (T_DIM*N_DIM)/32, H_DIM/32, 1);
  gemm(true ,true , Xq,  H_DIM,0, wW2, 0, b2, Xp, H_DIM, 0, 0, H_DIM, (T_DIM*N_DIM)/32, H_DIM/32, 1);
  gemm(false,false, Xp,  H_DIM,0, wWi_c,0, bi_c, Gi, G3H, 0, 0, H_DIM, (T_DIM*N_DIM)/32, 48, 1);

  // ========== main sequential scan: one fused kernel per step ===========
  hx_init<<<N_DIM/8, 256, 0, stream>>>(hx, p_arr, h_f, h_bf);   // h0 = hx[:,3:3+H]
  for (int t = 0; t < T_DIM; ++t)
    gru_step(wWh_c, bh_c, Gi + (long)t*N_DIM*G3H,
             Hsb + (long)t*N_DIM*H_DIM, H_DIM,
             out + (long)t*N_DIM*OUT_W + 3, OUT_W);

  // ========== batched heads over all (t, n) =============================
  // a-logits straight into d_out[:, 515:579], softmax in place
  gemm(false,false, Hsb, H_DIM, 0, wWa, 0, ba, out, OUT_W, 0, 515, H_DIM, (T_DIM*N_DIM)/32, NA_DIM/32, 1);
  softmax64<<<(T_DIM*N_DIM)/8, 256, 0, stream>>>(out, 515);
  // q = Hs @ Wl^T + bl (bf16)
  gemm(false,true , Hsb, H_DIM, 0, wWl, 0, bl, Qs, H2, 0, 0, H_DIM, (T_DIM*N_DIM)/32, H2/32, 1);
  // k[t,n,:] = K[n] @ q[t,n]  -> batched over n (z), logits into d_out[:,579:643]
  gemm(false,false, Qs, (long)N_DIM*H2, H2,
       Kbf, (long)L_DIM*H2, nullptr,
       out, (long)N_DIM*OUT_W, OUT_W, 579,
       H2, T_DIM/32, L_DIM/32, N_DIM);
  softmax64<<<(T_DIM*N_DIM)/8, 256, 0, stream>>>(out, 579);
  // v, A, P columns
  value_head<<<(T_DIM*N_DIM)/8, 256, 0, stream>>>(Wc, bc, out);
  ap_copy<<<(T_DIM*N_DIM)/256, 256, 0, stream>>>(inputs, out);

  // second output: hx_out[-1:]
  hipMemcpyAsync(out + (long)T_DIM*N_DIM*OUT_W, out + (long)(T_DIM-1)*N_DIM*OUT_W,
                 (size_t)N_DIM*OUT_W*sizeof(float), hipMemcpyDeviceToDevice, stream);
}
